// External_attention_12558484374220
// MI455X (gfx1250) — matrix operand order repaired and checked by reading, not compiled, not run on hardware
//
#include <hip/hip_runtime.h>
#include <hip/hip_bf16.h>

// ---------------------------------------------------------------------------
// External attention block for MI455X (gfx1250), wave32 + WMMA bf16 path.
// fp32 in HBM; tiles converted to bf16 (RNE) while staging to LDS; fp32
// accumulation via v_wmma_f32_16x16x32_bf16. Double-buffered LDS pipeline,
// float4 staging loads, branch-free main GEMMs (guards templated out).
// ---------------------------------------------------------------------------

typedef __attribute__((ext_vector_type(16))) __bf16          v16bf;
typedef __attribute__((ext_vector_type(8)))  float           v8f;
typedef __attribute__((ext_vector_type(8)))  unsigned short  u16x8;
typedef __attribute__((ext_vector_type(4)))  unsigned short  u16x4;
typedef __attribute__((ext_vector_type(4)))  float           f32x4;

union FragBF {
    v16bf v;
    u16x8 h[2];
};

__device__ __forceinline__ unsigned short f2bf(float f) {
    // round-to-nearest-even fp32 -> bf16
    unsigned int u = __float_as_uint(f);
    unsigned int r = u + 0x7FFFu + ((u >> 16) & 1u);
    return (unsigned short)(r >> 16);
}

#define TBM 128
#define TBN 128
#define TBK 32

// Out[b][m][n] = sum_k W[m][k] * X[b][k][n] (+ bias[m])
// W: [M,K] row-major, shared across batches. Requires K % 32 == 0, N % 128 == 0,
// n-grid exact. FULL_M => M % 128 == 0 (no row guards anywhere).
template <bool FULL_M>
__global__ __launch_bounds__(256) void gemm_wmma_bf16(
    const float* __restrict__ W, const float* __restrict__ X,
    float* __restrict__ Out, const float* __restrict__ bias,
    int M, int K, int N)
{
    // Double-buffered tiles. A row-major [row][k]; B transposed [col][k] so a
    // lane's 16 fragment elements are two contiguous 16B chunks (ds_load_b128).
    __shared__ unsigned short As[2][TBM][TBK];
    __shared__ unsigned short BsT[2][TBN][TBK];

    const int tid  = threadIdx.x;
    const int lane = tid & 31;
    const int wave = tid >> 5;        // 0..7
    const int wm   = wave >> 2;       // 0..1  -> 64 rows each
    const int wn   = wave & 3;        // 0..3  -> 32 cols each
    const int m0   = blockIdx.y * TBM;
    const int n0   = blockIdx.x * TBN;

    const float* Xb = X + (size_t)blockIdx.z * (size_t)K * (size_t)N;
    float*       Ob = Out + (size_t)blockIdx.z * (size_t)M * (size_t)N;

    const bool hi  = lane >= 16;
    const int  l15 = lane & 15;

    // Per-thread staging coordinates (float4 granularity).
    int ar[4], ac[4], br[4], bc[4];
    #pragma unroll
    for (int it = 0; it < 4; ++it) {
        int idx4 = it * 256 + tid;
        ar[it] = idx4 >> 3;          // 8 float4 per 32-wide A row
        ac[it] = (idx4 & 7) * 4;
        br[it] = idx4 >> 5;          // 32 float4 per 128-wide B row
        bc[it] = (idx4 & 31) * 4;
    }

    auto loadA = [&](int k0, f32x4 (&reg)[4]) {
        #pragma unroll
        for (int it = 0; it < 4; ++it) {
            int gm = m0 + ar[it];
            if (FULL_M || gm < M)
                reg[it] = *(const f32x4*)&W[(size_t)gm * K + (k0 + ac[it])];
            else
                reg[it] = f32x4{0.0f, 0.0f, 0.0f, 0.0f};
        }
    };
    auto loadB = [&](int k0, f32x4 (&reg)[4]) {
        #pragma unroll
        for (int it = 0; it < 4; ++it)
            reg[it] = *(const f32x4*)&Xb[(size_t)(k0 + br[it]) * N + (n0 + bc[it])];
    };
    auto stageA = [&](int buf, const f32x4 (&reg)[4]) {
        #pragma unroll
        for (int it = 0; it < 4; ++it) {
            u16x4 t = { f2bf(reg[it][0]), f2bf(reg[it][1]),
                        f2bf(reg[it][2]), f2bf(reg[it][3]) };
            *(u16x4*)&As[buf][ar[it]][ac[it]] = t;   // 8B aligned
        }
    };
    auto stageB = [&](int buf, const f32x4 (&reg)[4]) {
        #pragma unroll
        for (int it = 0; it < 4; ++it)
            #pragma unroll
            for (int t = 0; t < 4; ++t)
                BsT[buf][bc[it] + t][br[it]] = f2bf(reg[it][t]);  // transpose
    };

    v8f acc[4][2] = {};
    f32x4 pa[4], pb[4];

    loadA(0, pa); loadB(0, pb);
    stageA(0, pa); stageB(0, pb);
    __syncthreads();

    int buf = 0;
    const int kA = hi ? 8 : 0;    // A fragment chunk base (ISA 16-bit A 16x32)
    const int kB = hi ? 8 : 0;    // B fragment chunk base (same K order as A)

    for (int k0 = 0; k0 < K; k0 += TBK) {
        const bool more = (k0 + TBK) < K;
        if (more) { loadA(k0 + TBK, pa); loadB(k0 + TBK, pb); }

        FragBF a[4], b[2];
        #pragma unroll
        for (int mi = 0; mi < 4; ++mi) {
            int row = wm * 64 + mi * 16 + l15;
            a[mi].h[0] = *(const u16x8*)&As[buf][row][kA];       // K {0..7}+8*hi
            a[mi].h[1] = *(const u16x8*)&As[buf][row][kA + 16];  // K {16..23}+8*hi
        }
        #pragma unroll
        for (int ni = 0; ni < 2; ++ni) {
            int col = wn * 32 + ni * 16 + l15;
            b[ni].h[0] = *(const u16x8*)&BsT[buf][col][kB];       // K {0..7}+8*hi
            b[ni].h[1] = *(const u16x8*)&BsT[buf][col][kB + 16];  // K {16..23}+8*hi
        }

        #pragma unroll
        for (int mi = 0; mi < 4; ++mi)
            #pragma unroll
            for (int ni = 0; ni < 2; ++ni)
                acc[mi][ni] = __builtin_amdgcn_wmma_f32_16x16x32_bf16(
                    false, a[mi].v, false, b[ni].v,
                    (short)0, acc[mi][ni], false, false);

        if (more) { stageA(buf ^ 1, pa); stageB(buf ^ 1, pb); }
        __syncthreads();
        buf ^= 1;
    }

    // Epilogue: C/D layout — VGPR r holds M=r (lanes 0-15) / M=r+8 (lanes 16-31).
    #pragma unroll
    for (int mi = 0; mi < 4; ++mi) {
        int mbase = m0 + wm * 64 + mi * 16 + (hi ? 8 : 0);
        #pragma unroll
        for (int ni = 0; ni < 2; ++ni) {
            int n = n0 + wn * 32 + ni * 16 + l15;
            #pragma unroll
            for (int r = 0; r < 8; ++r) {
                int m = mbase + r;
                if (FULL_M || m < M) {
                    float v = acc[mi][ni][r];
                    if (bias) v += bias[m];
                    Ob[(size_t)m * N + n] = v;
                }
            }
        }
    }
}

// One block per (b,k) row: row max and sum of exp over n.
__global__ __launch_bounds__(256) void softmax_rowstats(
    const float* __restrict__ logits, float* __restrict__ rmax,
    float* __restrict__ rsum, int N)
{
    __shared__ float red[256];
    const float* p = logits + (size_t)blockIdx.x * N;
    float m = -3.0e38f;
    for (int i = threadIdx.x; i < N; i += 256) m = fmaxf(m, p[i]);
    red[threadIdx.x] = m; __syncthreads();
    for (int s = 128; s > 0; s >>= 1) {
        if (threadIdx.x < s) red[threadIdx.x] = fmaxf(red[threadIdx.x], red[threadIdx.x + s]);
        __syncthreads();
    }
    m = red[0]; __syncthreads();
    float sum = 0.0f;
    for (int i = threadIdx.x; i < N; i += 256) sum += expf(p[i] - m);
    red[threadIdx.x] = sum; __syncthreads();
    for (int s = 128; s > 0; s >>= 1) {
        if (threadIdx.x < s) red[threadIdx.x] += red[threadIdx.x + s];
        __syncthreads();
    }
    if (threadIdx.x == 0) { rmax[blockIdx.x] = m; rsum[blockIdx.x] = red[0]; }
}

// Per n column: softmax over n (via row stats) then L1 renorm over k. In-place.
__global__ __launch_bounds__(256) void attn_softmax_renorm(
    float* __restrict__ attn, const float* __restrict__ rmax,
    const float* __restrict__ rsum, int Kd, int N)
{
    const int b = blockIdx.y;
    const int n = blockIdx.x * 256 + threadIdx.x;
    float* base = attn + (size_t)b * Kd * N;
    const float* rm = rmax + b * Kd;
    const float* rs = rsum + b * Kd;
    float s = 0.0f;
    for (int k = 0; k < Kd; ++k)
        s += expf(base[(size_t)k * N + n] - rm[k]) / rs[k];
    float inv = 1.0f / (1e-9f + s);
    for (int k = 0; k < Kd; ++k) {
        size_t idx = (size_t)k * N + n;
        base[idx] = (expf(base[idx] - rm[k]) / rs[k]) * inv;
    }
}

// One block per channel: batch mean/var (biased) -> fused scale/shift.
__global__ __launch_bounds__(256) void bn_stats(
    const float* __restrict__ Z, const float* __restrict__ gamma,
    const float* __restrict__ beta, float* __restrict__ scale,
    float* __restrict__ shift, int B, int C, int N)
{
    __shared__ float r1[256], r2[256];
    const int o = blockIdx.x;
    float s = 0.0f, q = 0.0f;
    for (int b = 0; b < B; ++b) {
        const float* p = Z + ((size_t)b * C + o) * (size_t)N;
        for (int i = threadIdx.x; i < N; i += 256) {
            float v = p[i];
            s += v; q += v * v;
        }
    }
    r1[threadIdx.x] = s; r2[threadIdx.x] = q; __syncthreads();
    for (int st = 128; st > 0; st >>= 1) {
        if (threadIdx.x < st) {
            r1[threadIdx.x] += r1[threadIdx.x + st];
            r2[threadIdx.x] += r2[threadIdx.x + st];
        }
        __syncthreads();
    }
    if (threadIdx.x == 0) {
        float cnt  = (float)B * (float)N;
        float mean = r1[0] / cnt;
        float var  = r2[0] / cnt - mean * mean;
        float sc   = gamma[o] * rsqrtf(var + 1e-5f);
        scale[o] = sc;
        shift[o] = beta[o] - mean * sc;
    }
}

// out = relu(Z*scale[o] + shift[o] + x), in-place on d_out (holding Z).
__global__ __launch_bounds__(256) void bn_add_relu(
    float* __restrict__ out, const float* __restrict__ x,
    const float* __restrict__ scale, const float* __restrict__ shift,
    int C, int N, size_t total)
{
    size_t idx = (size_t)blockIdx.x * 256 + threadIdx.x;
    if (idx >= total) return;
    int o = (int)((idx / (size_t)N) % (size_t)C);
    float v = out[idx] * scale[o] + shift[o] + x[idx];
    out[idx] = v > 0.0f ? v : 0.0f;
}

extern "C" void kernel_launch(void* const* d_in, const int* in_sizes, int n_in,
                              void* d_out, int out_size, void* d_ws, size_t ws_size,
                              hipStream_t stream) {
    (void)in_sizes; (void)n_in; (void)out_size; (void)ws_size;
    const float* x       = (const float*)d_in[0];
    const float* conv1_w = (const float*)d_in[1];
    const float* conv1_b = (const float*)d_in[2];
    const float* lin0_w  = (const float*)d_in[3];
    const float* lin1_w  = (const float*)d_in[4];
    const float* conv2_w = (const float*)d_in[5];
    const float* gamma   = (const float*)d_in[6];
    const float* beta    = (const float*)d_in[7];
    float* out = (float*)d_out;

    const int B = 8, C = 512, Kd = 64, N = 128 * 128;   // N = 16384

    // workspace layout
    char*  ws    = (char*)d_ws;
    float* Y     = (float*)ws;                                   // B*C*N (reused)
    size_t offA  = (size_t)B * C * N * sizeof(float);
    float* attn  = (float*)(ws + offA);                          // B*Kd*N
    size_t offS  = offA + (size_t)B * Kd * N * sizeof(float);
    float* rmax  = (float*)(ws + offS);                          // B*Kd
    float* rsum  = rmax + B * Kd;                                // B*Kd
    float* scale = rsum + B * Kd;                                // C
    float* shift = scale + C;                                    // C

    dim3 blk(256);

    // 1) conv1 (1x1): Y = conv1_w @ x + conv1_b
    gemm_wmma_bf16<true><<<dim3(N / TBN, C / TBM, B), blk, 0, stream>>>(
        conv1_w, x, Y, conv1_b, C, C, N);

    // 2) lin0: logits = lin0_w @ Y   (M = 64 -> guarded variant)
    gemm_wmma_bf16<false><<<dim3(N / TBN, 1, B), blk, 0, stream>>>(
        lin0_w, Y, attn, nullptr, Kd, C, N);

    // 3) softmax over n + L1 renorm over k
    softmax_rowstats<<<dim3(B * Kd), blk, 0, stream>>>(attn, rmax, rsum, N);
    attn_softmax_renorm<<<dim3(N / 256, B), blk, 0, stream>>>(attn, rmax, rsum, Kd, N);

    // 4) lin1: Y = lin1_w @ attn   (K = 64; Y is dead, safe to reuse)
    gemm_wmma_bf16<true><<<dim3(N / TBN, C / TBM, B), blk, 0, stream>>>(
        lin1_w, attn, Y, nullptr, C, Kd, N);

    // 5) conv2: Z (stored in d_out) = conv2_w @ Y
    gemm_wmma_bf16<true><<<dim3(N / TBN, C / TBM, B), blk, 0, stream>>>(
        conv2_w, Y, out, nullptr, C, C, N);

    // 6) batch-norm stats -> fused scale/shift, then relu(Z_norm + x) in-place
    bn_stats<<<dim3(C), blk, 0, stream>>>(out, gamma, beta, scale, shift, B, C, N);
    size_t total = (size_t)B * C * N;
    bn_add_relu<<<dim3((unsigned)(total / 256)), blk, 0, stream>>>(
        out, x, scale, shift, C, N, total);
}
